// FFBrainNet_49821620634174
// MI455X (gfx1250) — compile-verified, matrix operand order checked by reading, not run on hardware
//
#include <hip/hip_runtime.h>

// Problem constants (from reference)
#define BATCH 8192
#define NIN   1024
#define WID   4096
#define MOUT  10
#define CAPK  512

// GEMM tiling
#define BM 128
#define BN 128
#define BK 32

typedef __attribute__((ext_vector_type(16))) __bf16 v16bf;
typedef __attribute__((ext_vector_type(8)))  float  v8f;

// CDNA5 async global->LDS path (guarded: falls back to load+ds_store if the
// toolchain doesn't declare the gfx1250 async builtins)
#if defined(__AMDGCN__) && \
    __has_builtin(__builtin_amdgcn_global_load_async_to_lds_b128) && \
    __has_builtin(__builtin_amdgcn_s_wait_asynccnt)
#define USE_ASYNC_LDS 1
// Builtin prototype (from hipcc diagnostic): param0 = gcc-vector int4 pointer
// in the global (__device__ / AS1) address space; param1 = LDS (AS3) side.
typedef int v4i_ __attribute__((vector_size(16)));
typedef __attribute__((address_space(1))) v4i_* gas_v4i;
typedef __attribute__((address_space(3))) v4i_* las_v4i;
#else
#define USE_ASYNC_LDS 0
#endif

__device__ __forceinline__ unsigned short f32_to_bf16(float f) {
    unsigned int u = __float_as_uint(f);
    u += 0x7FFFu + ((u >> 16) & 1u);          // round-to-nearest-even
    return (unsigned short)(u >> 16);
}
__device__ __forceinline__ float bf16_to_f32(unsigned short h) {
    return __uint_as_float(((unsigned int)h) << 16);
}

// ---------------------------------------------------------------------------
// Prep: out_bf16[i] = bf16(w[i] * m[i])   (m may be null -> plain convert)
// ---------------------------------------------------------------------------
__global__ __launch_bounds__(256) void mask_to_bf16(const float* __restrict__ w,
                                                    const float* __restrict__ m,
                                                    unsigned short* __restrict__ out,
                                                    size_t n) {
    size_t i = (size_t)blockIdx.x * blockDim.x + threadIdx.x;
    if (i < n) {
        float v = w[i];
        if (m) v *= m[i];
        out[i] = f32_to_bf16(v);
    }
}

__global__ __launch_bounds__(256) void mask_mul_f32(const float* __restrict__ w,
                                                    const float* __restrict__ m,
                                                    float* __restrict__ out,
                                                    size_t n) {
    size_t i = (size_t)blockIdx.x * blockDim.x + threadIdx.x;
    if (i < n) out[i] = w[i] * m[i];
}

// ---------------------------------------------------------------------------
// C[M,N] = relu(A[M,K] @ Bw[N,K]^T + bias[N])  -> bf16
// 256 threads (8 wave32), 128x128 block tile, BK=32, LDS double buffered.
// Each wave: 32x64 tile = 2x4 grid of 16x16 WMMA accumulators.
// Global->LDS fills use GLOBAL_LOAD_ASYNC_TO_LDS_B128 (ASYNCcnt) when
// available: no VGPR staging, overlapped with the 8 WMMAs per k-step.
// ---------------------------------------------------------------------------
__global__ __launch_bounds__(256) void gemm_bf16_relu(
    const unsigned short* __restrict__ A,   // [M,K] bf16
    const unsigned short* __restrict__ Bw,  // [N,K] bf16
    const float* __restrict__ bias,         // [N]
    unsigned short* __restrict__ C,         // [M,N] bf16
    int M, int N, int K)
{
    __shared__ __align__(64) unsigned short As[2][BM * BK];
    __shared__ __align__(64) unsigned short Bs[2][BN * BK];

    const int tid  = threadIdx.x;
    const int wave = tid >> 5;
    const int lane = tid & 31;

    const int rowBase = blockIdx.y * BM;
    const int colBase = blockIdx.x * BN;

    const int waveM = (wave & 3) * 32;   // 4 waves along M
    const int waveN = (wave >> 2) * 64;  // 2 waves along N

    v8f acc[2][4];
    for (int i = 0; i < 2; ++i)
        for (int j = 0; j < 4; ++j)
            for (int r = 0; r < 8; ++r)
                acc[i][j][r] = 0.0f;

    // global->LDS: 128x32 bf16 tile = 8KB; per thread 16 elems (2x b128)
    const int ldRow = tid >> 1;
    const int ldCol = (tid & 1) * 16;

    auto loadTile = [&](int buf, int k0) {
        const unsigned short* ag = A  + (size_t)(rowBase + ldRow) * K + k0 + ldCol;
        const unsigned short* bg = Bw + (size_t)(colBase + ldRow) * K + k0 + ldCol;
        unsigned short* la = &As[buf][ldRow * BK + ldCol];
        unsigned short* lb = &Bs[buf][ldRow * BK + ldCol];
#if USE_ASYNC_LDS
        __builtin_amdgcn_global_load_async_to_lds_b128((gas_v4i)(ag),     (las_v4i)(la),     0, 0);
        __builtin_amdgcn_global_load_async_to_lds_b128((gas_v4i)(ag + 8), (las_v4i)(la + 8), 0, 0);
        __builtin_amdgcn_global_load_async_to_lds_b128((gas_v4i)(bg),     (las_v4i)(lb),     0, 0);
        __builtin_amdgcn_global_load_async_to_lds_b128((gas_v4i)(bg + 8), (las_v4i)(lb + 8), 0, 0);
#else
        uint4 a0 = *(const uint4*)(ag);
        uint4 a1 = *(const uint4*)(ag + 8);
        *(uint4*)(la)     = a0;
        *(uint4*)(la + 8) = a1;
        uint4 b0 = *(const uint4*)(bg);
        uint4 b1 = *(const uint4*)(bg + 8);
        *(uint4*)(lb)     = b0;
        *(uint4*)(lb + 8) = b1;
        if (k0 + BK < K) {
            __builtin_prefetch(ag + BK, 0, 1);
            __builtin_prefetch(bg + BK, 0, 1);
        }
#endif
    };

    loadTile(0, 0);
#if USE_ASYNC_LDS
    __builtin_amdgcn_s_wait_asynccnt(0);
#endif
    __syncthreads();

    const int lr = lane & 15;         // row within 16-row fragment
    const int lk = (lane >> 4) * 16;  // K half selected by lane group

    for (int k0 = 0; k0 < K; k0 += BK) {
        const int buf = (k0 / BK) & 1;
        if (k0 + BK < K) loadTile(buf ^ 1, k0 + BK);

        v16bf afrag[2], bfrag[4];
        for (int i = 0; i < 2; ++i)
            afrag[i] = *(const v16bf*)&As[buf][(waveM + i * 16 + lr) * BK + lk];
        for (int j = 0; j < 4; ++j)
            bfrag[j] = *(const v16bf*)&Bs[buf][(waveN + j * 16 + lr) * BK + lk];

        for (int i = 0; i < 2; ++i)
            for (int j = 0; j < 4; ++j)
                acc[i][j] = __builtin_amdgcn_wmma_f32_16x16x32_bf16(
                    false, afrag[i], false, bfrag[j],
                    (short)0, acc[i][j], false, false);

#if USE_ASYNC_LDS
        __builtin_amdgcn_s_wait_asynccnt(0);
#endif
        __syncthreads();
    }

    // Epilogue: C/D 16x16 f32 layout: VGPR r -> M = r (+8 for lanes 16..31), N = lane%16
    const int nIdx = lane & 15;
    const int mAdd = (lane >> 4) * 8;
    for (int j = 0; j < 4; ++j) {
        const int col = colBase + waveN + j * 16 + nIdx;
        const float bv = bias[col];
        for (int i = 0; i < 2; ++i) {
            for (int r = 0; r < 8; ++r) {
                const int row = rowBase + waveM + i * 16 + mAdd + r;
                float v = acc[i][j][r] + bv;
                v = v > 0.0f ? v : 0.0f;
                C[(size_t)row * N + col] = f32_to_bf16(v);
            }
        }
    }
}

// ---------------------------------------------------------------------------
// k-winners-take-all, in place: keep top CAPK of WID bf16 values per row.
// Radix select on the bf16 bit pattern (values >= 0 after ReLU, so the
// unsigned 16-bit pattern is order-preserving). Two 8-bit histogram passes.
// ---------------------------------------------------------------------------
__global__ __launch_bounds__(256) void topk_inplace(unsigned short* __restrict__ act) {
    __shared__ __align__(16) unsigned short row[WID];
    __shared__ unsigned int hist[256];
    __shared__ unsigned int s_hi, s_rem, s_thresh;

    const int tid = threadIdx.x;
    unsigned short* g = act + (size_t)blockIdx.x * WID;

    for (int i = tid; i < WID / 8; i += 256)
        *(uint4*)&row[i * 8] = *(const uint4*)&g[i * 8];
    hist[tid] = 0;
    __syncthreads();

    for (int i = tid; i < WID; i += 256)
        atomicAdd(&hist[row[i] >> 8], 1u);
    __syncthreads();

    if (tid == 0) {
        unsigned int cum = 0;
        int b;
        for (b = 255; b > 0; --b) {
            if (cum + hist[b] >= CAPK) break;
            cum += hist[b];
        }
        s_hi  = (unsigned int)b;
        s_rem = CAPK - cum;
    }
    __syncthreads();
    const unsigned int hi = s_hi;

    hist[tid] = 0;
    __syncthreads();
    for (int i = tid; i < WID; i += 256) {
        unsigned short k = row[i];
        if ((unsigned int)(k >> 8) == hi) atomicAdd(&hist[k & 0xFF], 1u);
    }
    __syncthreads();

    if (tid == 0) {
        unsigned int rem = s_rem, cum = 0;
        int b;
        for (b = 255; b > 0; --b) {
            cum += hist[b];
            if (cum >= rem) break;
        }
        s_thresh = (hi << 8) | (unsigned int)b;
    }
    __syncthreads();

    const unsigned short th = (unsigned short)s_thresh;
    for (int i = tid; i < WID; i += 256) {
        unsigned short k = row[i];
        g[i] = (k >= th) ? k : (unsigned short)0;
    }
}

// ---------------------------------------------------------------------------
// logits = act[B,W] @ wm[M,W]^T + bias; softmax over M=10. One block per row.
// wm (160 KB) is L2-resident across all 8192 blocks.
// ---------------------------------------------------------------------------
__global__ __launch_bounds__(256) void out_softmax(
    const unsigned short* __restrict__ act,  // [B, WID] bf16
    const float* __restrict__ wm,            // [MOUT, WID]
    const float* __restrict__ obias,         // [MOUT]
    float* __restrict__ out)                 // [B, MOUT]
{
    __shared__ float red[256];
    __shared__ float logits[MOUT];
    const int tid = threadIdx.x;
    const unsigned short* a = act + (size_t)blockIdx.x * WID;

    float part[MOUT];
    for (int m = 0; m < MOUT; ++m) part[m] = 0.0f;

    for (int i = tid; i < WID; i += 256) {
        float av = bf16_to_f32(a[i]);
        if (av != 0.0f) {
            for (int m = 0; m < MOUT; ++m)
                part[m] += av * wm[m * WID + i];
        }
    }

    for (int m = 0; m < MOUT; ++m) {
        red[tid] = part[m];
        __syncthreads();
        for (int s = 128; s > 0; s >>= 1) {
            if (tid < s) red[tid] += red[tid + s];
            __syncthreads();
        }
        if (tid == 0) logits[m] = red[0] + obias[m];
        __syncthreads();
    }

    if (tid == 0) {
        float mx = logits[0];
        for (int m = 1; m < MOUT; ++m) mx = logits[m] > mx ? logits[m] : mx;
        float sum = 0.0f, e[MOUT];
        for (int m = 0; m < MOUT; ++m) { e[m] = __expf(logits[m] - mx); sum += e[m]; }
        float inv = 1.0f / sum;
        for (int m = 0; m < MOUT; ++m) out[(size_t)blockIdx.x * MOUT + m] = e[m] * inv;
    }
}

// ---------------------------------------------------------------------------
// Launch
// ---------------------------------------------------------------------------
extern "C" void kernel_launch(void* const* d_in, const int* in_sizes, int n_in,
                              void* d_out, int out_size, void* d_ws, size_t ws_size,
                              hipStream_t stream) {
    const float* x      = (const float*)d_in[0];
    const float* w_in   = (const float*)d_in[1];
    const float* gw1    = (const float*)d_in[2];
    const float* gw2    = (const float*)d_in[3];
    const float* b0     = (const float*)d_in[4];
    const float* b1     = (const float*)d_in[5];
    const float* b2     = (const float*)d_in[6];
    const float* w_out  = (const float*)d_in[7];
    const float* b_out  = (const float*)d_in[8];
    const float* m_in   = (const float*)d_in[9];
    const float* m1     = (const float*)d_in[10];
    const float* m2     = (const float*)d_in[11];
    const float* m_out  = (const float*)d_in[12];
    float* out = (float*)d_out;

    const size_t MB = 1024ull * 1024ull;
    char* ws = (char*)d_ws;
    unsigned short* xb    = (unsigned short*)(ws);             // 16 MB  (B*N bf16)
    unsigned short* wib   = (unsigned short*)(ws +  16 * MB);  //  8 MB  (W*N bf16)
    unsigned short* w1b   = (unsigned short*)(ws +  24 * MB);  // 32 MB  (W*W bf16)
    unsigned short* w2b   = (unsigned short*)(ws +  56 * MB);  // 32 MB  (W*W bf16)
    unsigned short* actA  = (unsigned short*)(ws +  88 * MB);  // 64 MB  (B*W bf16)
    unsigned short* actB  = (unsigned short*)(ws + 152 * MB);  // 64 MB  (B*W bf16)
    float*          wmout = (float*)(ws + 216 * MB);           // 160 KB (M*W f32)

    // --- prep: fold masks, convert to bf16 ---
    {
        size_t n = (size_t)BATCH * NIN;
        mask_to_bf16<<<(unsigned)((n + 255) / 256), 256, 0, stream>>>(x, nullptr, xb, n);
    }
    {
        size_t n = (size_t)WID * NIN;
        mask_to_bf16<<<(unsigned)((n + 255) / 256), 256, 0, stream>>>(w_in, m_in, wib, n);
    }
    {
        size_t n = (size_t)WID * WID;
        mask_to_bf16<<<(unsigned)((n + 255) / 256), 256, 0, stream>>>(gw1, m1, w1b, n);
        mask_to_bf16<<<(unsigned)((n + 255) / 256), 256, 0, stream>>>(gw2, m2, w2b, n);
    }
    {
        size_t n = (size_t)MOUT * WID;
        mask_mul_f32<<<(unsigned)((n + 255) / 256), 256, 0, stream>>>(w_out, m_out, wmout, n);
    }

    dim3 blk(256);

    // --- layer 1: [B,N] x [W,N]^T -> actA ---
    gemm_bf16_relu<<<dim3(WID / BN, BATCH / BM), blk, 0, stream>>>(
        xb, wib, b0, actA, BATCH, WID, NIN);
    topk_inplace<<<BATCH, blk, 0, stream>>>(actA);

    // --- layer 2: [B,W] x [W,W]^T -> actB ---
    gemm_bf16_relu<<<dim3(WID / BN, BATCH / BM), blk, 0, stream>>>(
        actA, w1b, b1, actB, BATCH, WID, WID);
    topk_inplace<<<BATCH, blk, 0, stream>>>(actB);

    // --- layer 3: [B,W] x [W,W]^T -> actA ---
    gemm_bf16_relu<<<dim3(WID / BN, BATCH / BM), blk, 0, stream>>>(
        actB, w2b, b2, actA, BATCH, WID, WID);
    topk_inplace<<<BATCH, blk, 0, stream>>>(actA);

    // --- head: logits + softmax ---
    out_softmax<<<BATCH, blk, 0, stream>>>(actA, wmout, b_out, out);
}
